// BOW_model_ta_20822001451179
// MI455X (gfx1250) — matrix operand order repaired and checked by reading, not compiled, not run on hardware
//
#include <hip/hip_runtime.h>
#include <hip/hip_bf16.h>
#include <math.h>

#define VV   100000
#define H    512
#define BB   4096
#define TT   524288
#define EPSF 1e-5f

typedef __attribute__((ext_vector_type(16))) _Float16 v16h;
typedef __attribute__((ext_vector_type(8)))  _Float16 v8h;
typedef __attribute__((ext_vector_type(4)))  _Float16 v4h;
typedef __attribute__((ext_vector_type(8)))  float    v8f;
typedef __attribute__((ext_vector_type(4)))  float    v4f;

// ---------------------------------------------------------------------------
// Stage 1: fused gather + segment-mean (seg_ids sorted) -> bow in f16.
// One block per segment; binary search for the token range.
// 256 threads = 2 token-parity halves x 128 float4 column-groups:
// per-lane b128 loads (512 B per wave-issue) to saturate HBM, LDS pair-reduce,
// then f16 convert + 64-bit store.
// ---------------------------------------------------------------------------
__device__ __forceinline__ int lower_bound_i32(const int* __restrict__ a, int n, int key) {
    int lo = 0, hi = n;
    while (lo < hi) {
        int mid = (lo + hi) >> 1;
        if (a[mid] < key) lo = mid + 1; else hi = mid;
    }
    return lo;
}

__global__ void seg_bow_f16_kernel(const int* __restrict__ tokens,
                                   const int* __restrict__ seg_ids,
                                   const float* __restrict__ emb,
                                   _Float16* __restrict__ bowh) {
    __shared__ v4f red[128];
    const int b   = blockIdx.x;        // segment id, 0..BB-1
    const int tx  = threadIdx.x;       // 256
    const int cg  = tx & 127;          // float4 column group: cols 4*cg..4*cg+3
    const int par = tx >> 7;           // token parity half (0 or 1)

    const int start = lower_bound_i32(seg_ids, TT, b);
    const int end   = lower_bound_i32(seg_ids, TT, b + 1);

    v4f acc = {0.0f, 0.0f, 0.0f, 0.0f};
    for (int s = start + par; s < end; s += 2) {
        const v4f row = *(const v4f*)(emb + (size_t)tokens[s] * H + cg * 4);
        acc += row;                    // global_load_b128 per lane, coalesced
    }
    if (par) red[cg] = acc;
    __syncthreads();
    if (!par) {
        acc += red[cg];
        const float inv = 1.0f / fmaxf((float)(end - start), 1.0f);
        const v4f scaled = acc * inv;
        const v4h h = __builtin_convertvector(scaled, v4h);
        *(v4h*)(bowh + (size_t)b * H + cg * 4) = h;   // 64-bit store
    }
}

// ---------------------------------------------------------------------------
// W1 f32 -> f16 (row-major [H,H]; rows of W1 are contiguous K for B = W1^T)
// ---------------------------------------------------------------------------
__global__ void cvt_w1_f16_kernel(const float* __restrict__ W1,
                                  _Float16* __restrict__ W1h) {
    const int i = blockIdx.x * blockDim.x + threadIdx.x;
    W1h[i] = (_Float16)W1[i];
}

// ---------------------------------------------------------------------------
// Stage 2: z = bow @ W1^T + b1 via v_wmma_f32_16x16x32_f16
// One wave -> one 16x16 tile of z; K-loop of 16 WMMA ops (K=512).
// A frag (16x32 f16): lane holds 16 contiguous K of row M=l16
//                     (lanes 0-15: K 0..15, lanes 16-31: K 16..31).
// B frag (32x16 f16): lane holds 16 contiguous K of column N=l16
//                     (lanes 0-15: K 0..15, lanes 16-31: K 16..31).
// C/D (16x16 f32):    VGPR r -> M = r (+8 for lanes 16-31), N = lane%16.
// ---------------------------------------------------------------------------
__global__ void gemm_wmma_kernel(const _Float16* __restrict__ A,   // bow f16 [BB,H]
                                 const _Float16* __restrict__ Wt,  // W1 f16 [H,H]
                                 const float* __restrict__ b1,
                                 float* __restrict__ z) {
    const int lane  = threadIdx.x & 31;
    const int wave  = threadIdx.x >> 5;
    const int gw    = blockIdx.x * 8 + wave;     // 8192 waves total
    const int tileM = gw >> 5;                   // 256 M-tiles
    const int tileN = gw & 31;                   // 32  N-tiles
    const int half  = lane >> 4;                 // 0: lanes 0-15, 1: lanes 16-31
    const int l16   = lane & 15;

    const _Float16* arow = A  + (size_t)(tileM * 16 + l16) * H;
    const _Float16* brow = Wt + (size_t)(tileN * 16 + l16) * H;
    const int koffA = half ? 16 : 0;
    const int koffB = half ? 16 : 0;

    v8f c = {};
#pragma unroll 4
    for (int k = 0; k < H; k += 32) {
        union { v16h v; v8h h[2]; } au;
        au.h[0] = *(const v8h*)(arow + k + koffA);        // K = k+koffA .. +7
        au.h[1] = *(const v8h*)(arow + k + koffA + 8);    // K = k+koffA+8 .. +15
        v16h bf = *(const v16h*)(brow + k + koffB);       // K = k+koffB .. +15
        c = __builtin_amdgcn_wmma_f32_16x16x32_f16(
                /*neg_a=*/false, au.v, /*neg_b=*/false, bf,
                /*c_mod=*/(short)0, c, /*reuse_a=*/false, /*reuse_b=*/false);
    }

    const int zcol = tileN * 16 + l16;
    const float bias = b1[zcol];
    const int zrow0 = tileM * 16 + (half ? 8 : 0);
#pragma unroll
    for (int r = 0; r < 8; ++r)
        z[(size_t)(zrow0 + r) * H + zcol] = c[r] + bias;
}

// ---------------------------------------------------------------------------
// Stage 3: per-column batch stats (mu, biased var) over B=4096 rows.
// Block = 32 cols x 8 row-threads; coalesced 32-float row chunks.
// ---------------------------------------------------------------------------
__global__ void bn_stats_kernel(const float* __restrict__ z,
                                float* __restrict__ mu,
                                float* __restrict__ var) {
    __shared__ float s_sum[8][32];
    __shared__ float s_sq[8][32];
    const int tx = threadIdx.x & 31;
    const int ty = threadIdx.x >> 5;
    const int col = blockIdx.x * 32 + tx;
    float s = 0.0f, q = 0.0f;
    for (int i = ty; i < BB; i += 8) {
        const float v = z[(size_t)i * H + col];
        s += v; q += v * v;
    }
    s_sum[ty][tx] = s; s_sq[ty][tx] = q;
    __syncthreads();
    if (ty == 0) {
#pragma unroll
        for (int r = 1; r < 8; ++r) { s += s_sum[r][tx]; q += s_sq[r][tx]; }
        const float m = s * (1.0f / (float)BB);
        mu[col]  = m;
        var[col] = q * (1.0f / (float)BB) - m * m;
    }
}

// ---------------------------------------------------------------------------
// Stage 4: fused BN (train-mode stats) + ReLU + dot(w2) + b2 -> logits
// One wave (32 lanes) per row; shuffle reduction (wave32).
// ---------------------------------------------------------------------------
__global__ void bn_relu_dot_kernel(const float* __restrict__ z,
                                   const float* __restrict__ mu,
                                   const float* __restrict__ var,
                                   const float* __restrict__ gamma,
                                   const float* __restrict__ beta,
                                   const float* __restrict__ w2,
                                   const float* __restrict__ b2,
                                   float* __restrict__ logits) {
    const int lane = threadIdx.x & 31;
    const int wave = threadIdx.x >> 5;
    const int i = blockIdx.x * 8 + wave;
    float acc = 0.0f;
    for (int j = lane; j < H; j += 32) {
        float zn = (z[(size_t)i * H + j] - mu[j]) * rsqrtf(var[j] + EPSF) * gamma[j] + beta[j];
        zn = fmaxf(zn, 0.0f);
        acc += zn * w2[j];
    }
#pragma unroll
    for (int off = 16; off > 0; off >>= 1)
        acc += __shfl_down(acc, off, 32);
    if (lane == 0) logits[i] = acc + b2[0];
}

// ---------------------------------------------------------------------------
// Stage 5: BCEWithLogits mean loss, numerically stable softplus.
// ---------------------------------------------------------------------------
__global__ void loss_kernel(const float* __restrict__ logits,
                            const float* __restrict__ t,
                            float* __restrict__ out) {
    __shared__ float sm[256];
    float acc = 0.0f;
    for (int i = threadIdx.x; i < BB; i += 256) {
        const float l  = logits[i];
        const float sp = fmaxf(l, 0.0f) + log1pf(expf(-fabsf(l)));
        acc += sp - t[i] * l;
    }
    sm[threadIdx.x] = acc;
    __syncthreads();
    for (int s = 128; s > 0; s >>= 1) {
        if (threadIdx.x < s) sm[threadIdx.x] += sm[threadIdx.x + s];
        __syncthreads();
    }
    if (threadIdx.x == 0) out[0] = sm[0] * (1.0f / (float)BB);
}

// ---------------------------------------------------------------------------
extern "C" void kernel_launch(void* const* d_in, const int* in_sizes, int n_in,
                              void* d_out, int out_size, void* d_ws, size_t ws_size,
                              hipStream_t stream) {
    (void)in_sizes; (void)n_in; (void)out_size; (void)ws_size;
    const int*   tokens  = (const int*)d_in[0];
    const int*   seg_ids = (const int*)d_in[1];
    const float* t       = (const float*)d_in[2];
    const float* emb     = (const float*)d_in[3];
    const float* W1      = (const float*)d_in[4];
    const float* b1      = (const float*)d_in[5];
    const float* gamma   = (const float*)d_in[6];
    const float* beta    = (const float*)d_in[7];
    const float* w2      = (const float*)d_in[8];
    const float* b2      = (const float*)d_in[9];

    float* out = (float*)d_out;        // out[0] = loss, out[1..4096] = logits
    char*  ws  = (char*)d_ws;

    // Workspace layout (all offsets 256B-aligned):
    _Float16* bowh = (_Float16*)ws;                                   // 4 MiB
    _Float16* w1h  = (_Float16*)(ws + (size_t)BB * H * 2);            // 512 KiB
    float*    z    = (float*)(ws + (size_t)BB * H * 2
                                 + (size_t)H * H * 2);                // 8 MiB
    float*    mu   = (float*)((char*)z + (size_t)BB * H * 4);         // 2 KiB
    float*    var  = mu + H;                                          // 2 KiB

    float* logits = out + 1;

    // 1) gather + segment mean -> bow (f16)
    seg_bow_f16_kernel<<<BB, 256, 0, stream>>>(tokens, seg_ids, emb, bowh);
    // 2) W1 -> f16
    cvt_w1_f16_kernel<<<(H * H) / 256, 256, 0, stream>>>(W1, w1h);
    // 3) z = bow @ W1^T + b1 (WMMA)
    gemm_wmma_kernel<<<(BB / 16) * (H / 16) / 8, 256, 0, stream>>>(bowh, w1h, b1, z);
    // 4) batch stats
    bn_stats_kernel<<<H / 32, 256, 0, stream>>>(z, mu, var);
    // 5) BN + ReLU + dot(w2) -> logits
    bn_relu_dot_kernel<<<BB / 8, 256, 0, stream>>>(z, mu, var, gamma, beta, w2, b2, logits);
    // 6) BCE loss
    loss_kernel<<<1, 256, 0, stream>>>(logits, t, out);
}
